// DynamicMQALNNLocalAttention_88983132438633
// MI455X (gfx1250) — matrix operand order repaired and checked by reading, not compiled, not run on hardware
//
#include <hip/hip_runtime.h>
#include <hip/hip_bf16.h>
#include <cstdint>
#include <cstddef>

// ---------------------------------------------------------------- types
typedef __attribute__((ext_vector_type(16))) __bf16 v16bf;
typedef __attribute__((ext_vector_type(8)))  __bf16 v8bf;
typedef __attribute__((ext_vector_type(4)))  __bf16 v4bf;
typedef __attribute__((ext_vector_type(8)))  float  v8f;
typedef __attribute__((ext_vector_type(4)))  unsigned int u32x4;
typedef __attribute__((ext_vector_type(8)))  int i32x8;
typedef __attribute__((ext_vector_type(4)))  int i32x4;

#define BN_SCALE 0.9999950000374997f   // 1/sqrt(1+1e-5)
#define S_LEN 2048
#define D_DIM 1024

#if defined(__has_builtin)
#if __has_builtin(__builtin_amdgcn_tensor_load_to_lds)
#define USE_TDM 1
#endif
#endif

static __device__ __forceinline__ float silu_f(float x) {
  return x / (1.0f + __expf(-x));
}
static __device__ __forceinline__ float gelu_f(float x) {
  return 0.5f * x * (1.0f + erff(x * 0.70710678118654752f));
}
static __device__ __forceinline__ v16bf ld16(const __bf16* p0, const __bf16* p1) {
  v8bf lo = *(const v8bf*)p0;
  v8bf hi = *(const v8bf*)p1;
  return __builtin_shufflevector(lo, hi, 0,1,2,3,4,5,6,7,8,9,10,11,12,13,14,15);
}
static __device__ __forceinline__ v8f wmma_bf16(v16bf a, v16bf b, v8f c) {
  return __builtin_amdgcn_wmma_f32_16x16x32_bf16(false, a, false, b, (short)0, c,
                                                 false, false);
}

// XOR butterfly within 16-lane groups via ds_swizzle (1 instruction, no clamp
// VALU): group-of-32 mode, xor_mask=M, or_mask=0, and_mask=0x1f.
template <int M>
static __device__ __forceinline__ float swz_xor(float v) {
  return __int_as_float(
      __builtin_amdgcn_ds_swizzle(__float_as_int(v), (M << 10) | 0x1f));
}
static __device__ __forceinline__ float redmax16(float v) {
  v = fmaxf(v, swz_xor<1>(v)); v = fmaxf(v, swz_xor<2>(v));
  v = fmaxf(v, swz_xor<4>(v)); v = fmaxf(v, swz_xor<8>(v));
  return v;
}
static __device__ __forceinline__ float redsum16(float v) {
  v += swz_xor<1>(v); v += swz_xor<2>(v);
  v += swz_xor<4>(v); v += swz_xor<8>(v);
  return v;
}

#ifdef USE_TDM
// ---------------------------------------------------------------- TDM 2-D tile load
// Loads a (rows x 32) bf16 tile (row stride ld_elems in global) into LDS at
// lds_off, padding each 64B row with 32B so the LDS row stride is 96B
// (matches the [.][48] __bf16 LDS arrays). Wave-level; caller waits TENSORcnt.
static __device__ __forceinline__ void tdm_load_tile(unsigned lds_off,
                                                     const __bf16* gsrc,
                                                     int rows, int ld_elems) {
  unsigned long long ga = (unsigned long long)(uintptr_t)gsrc;
  u32x4 g0;
  g0[0] = 1u;                                     // count=1, user mode
  g0[1] = lds_off;                                // lds_addr
  g0[2] = (unsigned)ga;                           // global_addr[31:0]
  g0[3] = (unsigned)(ga >> 32) | (2u << 30);      // global_addr[56:32] | type=2
  i32x8 g1;
  g1[0] = (1 << 16)        // data_size = 2 bytes
        | (1 << 20)        // pad_enable
        | (3 << 22)        // pad_interval: every 64 bytes
        | (7 << 25);       // pad_amount: 8 dwords = 32 bytes
  g1[1] = (int)(32u << 16);                       // tensor_dim0[15:0] = 32
  g1[2] = (int)(((unsigned)rows) << 16);          // tensor_dim1[15:0] = rows
  g1[3] = (int)(32u << 16);                       // tile_dim0 = 32
  g1[4] = rows;                                   // tile_dim1 = rows, tile_dim2=0
  g1[5] = ld_elems;                               // tensor_dim0_stride lo32
  g1[6] = 0;
  g1[7] = 0;
  i32x4 z4 = {0, 0, 0, 0};
#if __clang_major__ >= 23
  i32x8 z8 = {0, 0, 0, 0, 0, 0, 0, 0};
  __builtin_amdgcn_tensor_load_to_lds(g0, g1, z4, z4, z8, 0);
#else
  __builtin_amdgcn_tensor_load_to_lds(g0, g1, z4, z4, 0);
#endif
}
#endif

// ---------------------------------------------------------------- f32 -> bf16
__global__ void f2bf_kernel(const float* __restrict__ s,
                            __bf16* __restrict__ d, int n4) {
  int i = blockIdx.x * blockDim.x + threadIdx.x;
  if (i >= n4) return;
  float4 f = ((const float4*)s)[i];
  v4bf p; p[0] = (__bf16)f.x; p[1] = (__bf16)f.y;
          p[2] = (__bf16)f.z; p[3] = (__bf16)f.w;
  ((v4bf*)d)[i] = p;
}

// ---------------------------------------------------------------- GEMM (NT, bf16 in)
// C(M,N) = act( A(M,K) * B(N,K)^T [+ C if accum] [+ bias] )
// ACT: 0=none, 1=silu, 2=gelu(x*BN_SCALE)
// BFMODE: 0=f32 only, 1=bf16 row-major only, 2=bf16 transposed only, 3=f32+bf16
// TDM path: double-buffered LDS tiles; wave0 issues next-tile DMA while all
// waves run WMMAs on the current tile (DMA/compute overlap).
template <int ACT, int BFMODE>
__global__ __launch_bounds__(256) void gemm_nt_kernel(
    const __bf16* __restrict__ A, int lda,
    const __bf16* __restrict__ Bw, int ldb,
    const float* __restrict__ bias,
    float* __restrict__ C, int ldc,
    __bf16* __restrict__ Cbf, int ldcbf,
    int K, int accum) {
#ifdef USE_TDM
  __shared__ __align__(16) __bf16 As[2][128][48];
  __shared__ __align__(16) __bf16 Bs[2][64][48];
#else
  __shared__ __align__(16) __bf16 As[1][128][48];
  __shared__ __align__(16) __bf16 Bs[1][64][48];
#endif
  const int tid = threadIdx.x;
  const int rowBase = blockIdx.y * 128;
  const int colBase = blockIdx.x * 64;
  const int wave = tid >> 5;
  const int lane = tid & 31;
  const int m16  = lane & 15;
  const int gh   = lane >> 4;

  v8f acc[4] = {};
  int ib = 0;

#ifdef USE_TDM
  const unsigned ldsA0 = (unsigned)(uintptr_t)&As[0][0][0];
  const unsigned ldsA1 = (unsigned)(uintptr_t)&As[1][0][0];
  const unsigned ldsB0 = (unsigned)(uintptr_t)&Bs[0][0][0];
  const unsigned ldsB1 = (unsigned)(uintptr_t)&Bs[1][0][0];
  if (wave == 0) {                       // prologue: fill buffer 0
    tdm_load_tile(ldsA0, A + (size_t)rowBase * lda, 128, lda);
    tdm_load_tile(ldsB0, Bw + (size_t)colBase * ldb, 64, ldb);
  }
#endif

  for (int kk = 0; kk < K; kk += 32) {
#ifdef USE_TDM
    if (wave == 0) __builtin_amdgcn_s_wait_tensorcnt(0);
    __syncthreads();                     // current buffer ready for all waves
    if (wave == 0 && kk + 32 < K) {      // overlap: DMA next tile into other buf
      tdm_load_tile(ib ? ldsA0 : ldsA1, A + (size_t)rowBase * lda + kk + 32, 128, lda);
      tdm_load_tile(ib ? ldsB0 : ldsB1, Bw + (size_t)colBase * ldb + kk + 32, 64, ldb);
    }
#else
#pragma unroll
    for (int i = 0; i < 2; ++i) {        // A tile 128x32 bf16, 512 x v8bf
      int idx = tid + i * 256;
      int r = idx >> 2;
      int c = (idx & 3) << 3;
      *(v8bf*)&As[0][r][c] = *(const v8bf*)(A + (size_t)(rowBase + r) * lda + kk + c);
    }
    {                                    // B tile 64x32 bf16, 256 x v8bf
      int r = tid >> 2;
      int c = (tid & 3) << 3;
      *(v8bf*)&Bs[0][r][c] = *(const v8bf*)(Bw + (size_t)(colBase + r) * ldb + kk + c);
    }
    __syncthreads();
#endif
    const int arow = wave * 16 + m16;
    v16bf afrag = ld16(&As[ib][arow][gh * 16], &As[ib][arow][gh * 16 + 8]);
#pragma unroll
    for (int nt = 0; nt < 4; ++nt) {
      v16bf bfrag = ld16(&Bs[ib][nt * 16 + m16][gh * 16],
                         &Bs[ib][nt * 16 + m16][gh * 16 + 8]);
      acc[nt] = wmma_bf16(afrag, bfrag, acc[nt]);
    }
    __syncthreads();                     // reads done before buffer is re-filled
#ifdef USE_TDM
    ib ^= 1;
#endif
  }

#pragma unroll
  for (int nt = 0; nt < 4; ++nt) {
#pragma unroll
    for (int r = 0; r < 8; ++r) {
      const int row = rowBase + wave * 16 + gh * 8 + r;
      const int col = colBase + nt * 16 + m16;
      float x = acc[nt][r];
      if (accum) x += C[(size_t)row * ldc + col];
      if (bias)  x += bias[col];
      if (ACT == 1) x = silu_f(x);
      else if (ACT == 2) x = gelu_f(x * BN_SCALE);
      if (BFMODE == 0 || BFMODE == 3) C[(size_t)row * ldc + col] = x;
      if (BFMODE == 1 || BFMODE == 3) Cbf[(size_t)row * ldcbf + col] = (__bf16)x;
      if (BFMODE == 2)                Cbf[(size_t)col * ldcbf + row] = (__bf16)x;
    }
  }
}

// ---------------------------------------------------------------- depthwise k=3
__global__ void dwconv3_kernel(const __bf16* __restrict__ h,
                               const float* __restrict__ w,
                               __bf16* __restrict__ y, int S, int C) {
  int i = blockIdx.x * blockDim.x + threadIdx.x;
  if (i >= S * C) return;
  int t = i / C, c = i - t * C;
  float a = (t > 0)     ? (float)h[i - C] : 0.0f;
  float b = (float)h[i];
  float d = (t < S - 1) ? (float)h[i + C] : 0.0f;
  float v = (a * w[c * 3 + 0] + b * w[c * 3 + 1] + d * w[c * 3 + 2]) * BN_SCALE;
  y[i] = (__bf16)silu_f(v);
}

// ---------------------------------------------------------------- per-channel scale + silu
__global__ void chan_scale_silu_kernel(__bf16* __restrict__ x,
                                       const float* __restrict__ w, int total) {
  int i = blockIdx.x * blockDim.x + threadIdx.x;
  if (i >= total) return;
  x[i] = (__bf16)silu_f((float)x[i] * w[i & (D_DIM - 1)] * BN_SCALE);
}

// ---------------------------------------------------------------- window predictor reduce
__global__ void zero_scalar_kernel(float* a) { *a = 0.0f; }

__global__ void wred_kernel(const __bf16* __restrict__ g,
                            const float* __restrict__ wp2,
                            float* __restrict__ accum, int total) {
  float s = 0.0f;
  for (int i = blockIdx.x * blockDim.x + threadIdx.x; i < total;
       i += gridDim.x * blockDim.x)
    s += (float)g[i] * wp2[i & (D_DIM - 1)];
  __shared__ float red[256];
  red[threadIdx.x] = s;
  __syncthreads();
  for (int st = 128; st; st >>= 1) {
    if (threadIdx.x < st) red[threadIdx.x] += red[threadIdx.x + st];
    __syncthreads();
  }
  if (threadIdx.x == 0) atomicAdd(accum, red[0]);
}

__global__ void win_final_kernel(const float* __restrict__ accum,
                                 int* __restrict__ win, int S) {
  float ratio = accum[0] * BN_SCALE / (float)S;
  float wf = 3.0f + ratio * 29.0f;             // MINW + ratio*(MAXW-MINW)
  int w = (int)truncf(wf);
  if (w > S) w = S;
  *win = w;
}

// ---------------------------------------------------------------- flash MQA attention
__global__ __launch_bounds__(32) void attn_kernel(
    const __bf16* __restrict__ qbf,   // S x D   (bf16)
    const __bf16* __restrict__ kbf,   // S x 64  (bf16)
    const __bf16* __restrict__ vtbf,  // 64 x S  (bf16, transposed)
    __bf16* __restrict__ attn,        // S x D   (bf16)
    const int* __restrict__ winp, int S) {
  __shared__ __align__(16) __bf16 Ps[16][48];
  const int lane = threadIdx.x;
  const int m = lane & 15, g = lane >> 4;
  const int qb = blockIdx.x, head = blockIdx.y;
  const int window = *winp;
  const bool full = (window >= S);
  const int hw = window >> 1;

  const __bf16* qrow = qbf + (size_t)(qb * 16 + m) * D_DIM + head * 64;
  v16bf aq0 = ld16(qrow + g * 16,      qrow + g * 16 + 8);
  v16bf aq1 = ld16(qrow + 32 + g * 16, qrow + 40 + g * 16);

  float mrow[8], lrow[8];
  v8f o[4] = {};
#pragma unroll
  for (int r = 0; r < 8; ++r) { mrow[r] = -3.0e38f; lrow[r] = 0.0f; }

  const int lo_blk = qb * 16 - hw;
  const int hi_blk = qb * 16 + 15 + hw;

  for (int kv0 = 0; kv0 < S; kv0 += 32) {
    if (!full && (kv0 > hi_blk || kv0 + 31 < lo_blk)) continue;
    if (kv0 + 32 < S)
      __builtin_prefetch(kbf + (size_t)(kv0 + 32 + m) * 64, 0, 1);

    // scores: 16x32 = two 16x16 tiles, K(=hd)=64 in two WMMA steps
    v8f s0 = {}, s1 = {};
    const __bf16* kr0 = kbf + (size_t)(kv0 + m) * 64;
    const __bf16* kr1 = kbf + (size_t)(kv0 + 16 + m) * 64;
    {
      v16bf b00 = ld16(kr0 + g * 16,      kr0 + g * 16 + 8);
      v16bf b10 = ld16(kr1 + g * 16,      kr1 + g * 16 + 8);
      s0 = wmma_bf16(aq0, b00, s0);
      s1 = wmma_bf16(aq0, b10, s1);
      v16bf b01 = ld16(kr0 + 32 + g * 16, kr0 + 40 + g * 16);
      v16bf b11 = ld16(kr1 + 32 + g * 16, kr1 + 40 + g * 16);
      s0 = wmma_bf16(aq1, b01, s0);
      s1 = wmma_bf16(aq1, b11, s1);
    }

    const int col0 = kv0 + m, col1 = kv0 + 16 + m;
#pragma unroll
    for (int r = 0; r < 8; ++r) {
      const int row = qb * 16 + g * 8 + r;
      float x0 = s0[r] * 0.125f, x1 = s1[r] * 0.125f;  // 1/sqrt(64)
      if (!full) {
        if (col0 < row - hw || col0 > row + hw) x0 = -1.0e9f;
        if (col1 < row - hw || col1 > row + hw) x1 = -1.0e9f;
      }
      const float vmax = redmax16(fmaxf(x0, x1));
      const float mnew = fmaxf(mrow[r], vmax);
      const float sc = __expf(mrow[r] - mnew);
      const float p0 = __expf(x0 - mnew), p1 = __expf(x1 - mnew);
      const float rs = redsum16(p0 + p1);
      lrow[r] = lrow[r] * sc + rs;
      mrow[r] = mnew;
#pragma unroll
      for (int t = 0; t < 4; ++t) o[t][r] = o[t][r] * sc;
      Ps[g * 8 + r][m]      = (__bf16)p0;
      Ps[g * 8 + r][16 + m] = (__bf16)p1;
    }
    __syncthreads();                       // C-layout -> A-layout via LDS
    v16bf ap = ld16(&Ps[m][g * 16], &Ps[m][g * 16 + 8]);
#pragma unroll
    for (int t = 0; t < 4; ++t) {
      const __bf16* vr = vtbf + (size_t)(t * 16 + m) * S + kv0 + g * 16;
      v16bf bv = ld16(vr, vr + 8);
      o[t] = wmma_bf16(ap, bv, o[t]);
    }
    __syncthreads();
  }

#pragma unroll
  for (int t = 0; t < 4; ++t)
#pragma unroll
    for (int r = 0; r < 8; ++r) {
      const int row = qb * 16 + g * 8 + r;
      const int col = head * 64 + t * 16 + m;
      attn[(size_t)row * D_DIM + col] = (__bf16)(o[t][r] / lrow[r]);
    }
}

// ---------------------------------------------------------------- gated combine
__global__ void combine_kernel(const float* __restrict__ q,
                               const float* __restrict__ a,
                               const float* __restrict__ gate,
                               float* __restrict__ out, int total) {
  int i = blockIdx.x * blockDim.x + threadIdx.x;
  if (i >= total) return;
  float g = gate[i];
  out[i] = q[i] * g + a[i] * (1.0f - g);
}

// ---------------------------------------------------------------- RMSNorm per row
__global__ __launch_bounds__(256) void rms_kernel(const float* __restrict__ x,
                                                  const float* __restrict__ w,
                                                  float* __restrict__ y, int D) {
  const int row = blockIdx.x;
  const float* xr = x + (size_t)row * D;
  float ss = 0.0f;
  for (int c = threadIdx.x; c < D; c += 256) { float v = xr[c]; ss += v * v; }
  __shared__ float red[256];
  red[threadIdx.x] = ss;
  __syncthreads();
  for (int st = 128; st; st >>= 1) {
    if (threadIdx.x < st) red[threadIdx.x] += red[threadIdx.x + st];
    __syncthreads();
  }
  __shared__ float rms;
  if (threadIdx.x == 0) rms = rsqrtf(red[0] / (float)D + 1e-6f);
  __syncthreads();
  for (int c = threadIdx.x; c < D; c += 256)
    y[(size_t)row * D + c] = xr[c] * rms * w[c];
}

// ---------------------------------------------------------------- column-mean subtract
__global__ void colmean_kernel(const float* __restrict__ h,
                               float* __restrict__ out, int S, int D) {
  const int c = blockIdx.x * blockDim.x + threadIdx.x;
  if (c >= D) return;
  float s = 0.0f;
  for (int t = 0; t < S; ++t) s += h[(size_t)t * D + c];
  const float mean = s / (float)S;
  for (int t = 0; t < S; ++t)
    out[(size_t)t * D + c] = h[(size_t)t * D + c] - mean;
}

// ---------------------------------------------------------------- launch
extern "C" void kernel_launch(void* const* d_in, const int* in_sizes, int n_in,
                              void* d_out, int out_size, void* d_ws, size_t ws_size,
                              hipStream_t stream) {
  (void)in_sizes; (void)n_in; (void)out_size; (void)ws_size;
  const float* query  = (const float*)d_in[0];
  const float* key_in = (const float*)d_in[1];
  const float* val_in = (const float*)d_in[2];
  const float* w_e1   = (const float*)d_in[3];
  const float* w_dw1  = (const float*)d_in[4];
  const float* w_p1   = (const float*)d_in[5];
  const float* w_e2   = (const float*)d_in[6];
  const float* w_dw2  = (const float*)d_in[7];
  const float* w_p2   = (const float*)d_in[8];
  const float* wq = (const float*)d_in[9];  const float* bq = (const float*)d_in[10];
  const float* wk = (const float*)d_in[11]; const float* bk = (const float*)d_in[12];
  const float* wv = (const float*)d_in[13]; const float* bv = (const float*)d_in[14];
  const float* wo = (const float*)d_in[15]; const float* bo = (const float*)d_in[16];
  const float* wg = (const float*)d_in[17]; const float* bg = (const float*)d_in[18];
  const float* rms_w = (const float*)d_in[19];
  float* out = (float*)d_out;

  const int S = S_LEN, D = D_DIM;
  const size_t KB = 1024, MB = KB * KB;
  char* ws = (char*)d_ws;
  // persistent bf16 weights
  __bf16* we1b  = (__bf16*)(ws + 0);                  // 8MB
  __bf16* wp1b  = (__bf16*)(ws + 8 * MB);             // 2MB
  __bf16* we2b  = (__bf16*)(ws + 10 * MB);            // 512KB
  __bf16* wqb   = (__bf16*)(ws + 10 * MB + 512 * KB); // 2MB
  __bf16* wkb   = (__bf16*)(ws + 12 * MB + 512 * KB); // 128KB
  __bf16* wvb   = (__bf16*)(ws + 12 * MB + 640 * KB); // 128KB
  __bf16* wob   = (__bf16*)(ws + 12 * MB + 768 * KB); // 2MB
  __bf16* wgb   = (__bf16*)(ws + 14 * MB + 768 * KB); // 4MB
  __bf16* qryb  = (__bf16*)(ws + 19 * MB);            // 4MB
  __bf16* keyb  = (__bf16*)(ws + 23 * MB);            // 4MB
  __bf16* valb  = (__bf16*)(ws + 27 * MB);            // 4MB
  // intermediates (with reuse)
  __bf16* h1b   = (__bf16*)(ws + 31 * MB);            // 16MB (e1 out)
  __bf16* h1c   = (__bf16*)(ws + 47 * MB);            // 16MB (dwconv out)
  __bf16* h2b   = (__bf16*)(ws + 31 * MB);            // 1MB  (reuses h1b)
  __bf16* gb    = (__bf16*)(ws + 32 * MB);            // 4MB
  __bf16* qbf   = (__bf16*)(ws + 36 * MB);            // 4MB
  __bf16* kbf   = (__bf16*)(ws + 40 * MB);            // 256KB
  __bf16* vtbf  = (__bf16*)(ws + 40 * MB + 256 * KB); // 256KB
  __bf16* attnb = (__bf16*)(ws + 41 * MB);            // 4MB
  float*  attnp = (float*)(ws + 45 * MB);             // 8MB (reuses h1c)
  __bf16* attnpb= (__bf16*)(ws + 53 * MB);            // 4MB
  float*  gate  = (float*)(ws + 57 * MB);             // 8MB
  float*  comb  = (float*)(ws + 65 * MB);             // 8MB
  float*  hbuf  = (float*)(ws + 73 * MB);             // 8MB
  float*  accum = (float*)(ws + 81 * MB);
  int*    win   = (int*)(ws + 81 * MB + 64);

  dim3 blk(256);
  // 0. pre-convert operands to bf16 (raw-copyable by TDM, half the tile bytes)
  auto cvt = [&](const float* s, __bf16* d, int n) {
    f2bf_kernel<<<(n / 4 + 255) / 256, blk, 0, stream>>>(s, d, n / 4);
  };
  cvt(w_e1, we1b, 4 * D * D); cvt(w_p1, wp1b, (D / 4) * 4 * D);
  cvt(w_e2, we2b, D * (D / 4)); cvt(wq, wqb, D * D);
  cvt(wk, wkb, 64 * D); cvt(wv, wvb, 64 * D);
  cvt(wo, wob, D * D); cvt(wg, wgb, D * 2 * D);
  cvt(query, qryb, S * D); cvt(key_in, keyb, S * D); cvt(val_in, valb, S * D);

  // 1. h1 = silu(query @ w_e1^T)                       (2048 x 4096, K=1024)
  gemm_nt_kernel<1, 1><<<dim3(4096 / 64, S / 128), blk, 0, stream>>>(
      qryb, D, we1b, D, nullptr, nullptr, 0, h1b, 4096, D, 0);
  // 2. depthwise conv k=3 + BN scale + silu
  dwconv3_kernel<<<(S * 4096 + 255) / 256, blk, 0, stream>>>(h1b, w_dw1, h1c, S, 4096);
  // 3. h2 = gelu(BN * (h1c @ w_p1^T))                  (2048 x 256, K=4096)
  gemm_nt_kernel<2, 1><<<dim3(256 / 64, S / 128), blk, 0, stream>>>(
      h1c, 4096, wp1b, 4096, nullptr, nullptr, 0, h2b, 256, 4096, 0);
  // 4. g = silu(h2 @ w_e2^T)                           (2048 x 1024, K=256)
  gemm_nt_kernel<1, 1><<<dim3(D / 64, S / 128), blk, 0, stream>>>(
      h2b, 256, we2b, 256, nullptr, nullptr, 0, gb, D, 256, 0);
  // 5. g = silu(g * w_dw2 * BN)
  chan_scale_silu_kernel<<<(S * D + 255) / 256, blk, 0, stream>>>(gb, w_dw2, S * D);
  // 6. window = trunc(3 + mean(g @ w_p2^T * BN) * 29), clamp to S
  zero_scalar_kernel<<<1, 1, 0, stream>>>(accum);
  wred_kernel<<<256, blk, 0, stream>>>(gb, w_p2, accum, S * D);
  win_final_kernel<<<1, 1, 0, stream>>>(accum, win, S);
  // 7. q (bf16), k (bf16), v^T (bf16)
  gemm_nt_kernel<0, 1><<<dim3(D / 64, S / 128), blk, 0, stream>>>(
      qryb, D, wqb, D, bq, nullptr, 0, qbf, D, D, 0);
  gemm_nt_kernel<0, 1><<<dim3(1, S / 128), blk, 0, stream>>>(
      keyb, D, wkb, D, bk, nullptr, 0, kbf, 64, D, 0);
  gemm_nt_kernel<0, 2><<<dim3(1, S / 128), blk, 0, stream>>>(
      valb, D, wvb, D, bv, nullptr, 0, vtbf, S, D, 0);
  // 8. flash MQA attention with dynamic window
  attn_kernel<<<dim3(S / 16, 16), dim3(32), 0, stream>>>(qbf, kbf, vtbf, attnb, win, S);
  // 9. attn = attnb @ wo^T + bo   (f32 + bf16 outputs)
  gemm_nt_kernel<0, 3><<<dim3(D / 64, S / 128), blk, 0, stream>>>(
      attnb, D, wob, D, bo, attnp, D, attnpb, D, D, 0);
  // 10. gate = silu([query, attn] @ wg^T + bg)  -- split wg (D x 2D), two passes
  gemm_nt_kernel<0, 0><<<dim3(D / 64, S / 128), blk, 0, stream>>>(
      qryb, D, wgb, 2 * D, nullptr, gate, D, nullptr, 0, D, 0);
  gemm_nt_kernel<1, 0><<<dim3(D / 64, S / 128), blk, 0, stream>>>(
      attnpb, D, wgb + D, 2 * D, bg, gate, D, nullptr, 0, D, 1);
  // 11. out = query*gate + attn*(1-gate)
  combine_kernel<<<(S * D + 255) / 256, blk, 0, stream>>>(query, attnp, gate, comb, S * D);
  // 12. RMSNorm per row
  rms_kernel<<<S, blk, 0, stream>>>(comb, rms_w, hbuf, D);
  // 13. subtract per-feature mean over sequence -> d_out
  colmean_kernel<<<(D + 255) / 256, blk, 0, stream>>>(hbuf, out, S, D);
}
